// TritonINT4Linear_68951404970061
// MI455X (gfx1250) — matrix operand order repaired and checked by reading, not compiled, not run on hardware
//
#include <hip/hip_runtime.h>
#include <stdint.h>

// ---------------------------------------------------------------------------
// INT4 group-quantized GEMM for MI455X (gfx1250, wave32, WMMA).
//   out[M,N](f16) = x[M,K](f16) @ dequant(w_packed[K,N/2] u8-nibbles), f32 acc
//   dequant: w = (nibble - zero[k/G, n]) * scale[k/G, n],  G = 128
//
// Block: 128(M) x 128(N), 8 waves, wave = 32x64 = 2x4 WMMA tiles.
// K-step 64 (2 WMMA k-slices), double-buffered LDS, software pipelined:
//   - x tile staged via global_load_async_to_lds_b128 (ASYNCcnt)
//   - weights pipelined through registers; nibble pairs gathered with
//     v_perm_b32 + 0x6400-bias trick, dequantized with packed f16 math and
//     stored as ds_store_b32 into a column-major LDS tile so B fragments are
//     two contiguous ds_load_b128.
// ---------------------------------------------------------------------------

typedef _Float16 v16h __attribute__((ext_vector_type(16)));
typedef _Float16 v8h  __attribute__((ext_vector_type(8)));
typedef _Float16 h2   __attribute__((ext_vector_type(2)));
typedef float    v8f  __attribute__((ext_vector_type(8)));
typedef int      i4v  __attribute__((ext_vector_type(4)));
typedef uint32_t u4v  __attribute__((ext_vector_type(4)));

#define BLOCK_M 128
#define BLOCK_N 128
#define BLOCK_K 64
#define A_PAD   72    // 64 K-halves + 8 -> 144B rows (16B aligned, bank-safe)
#define B_PAD   72

#if __has_builtin(__builtin_amdgcn_global_load_async_to_lds_b128)
typedef __attribute__((address_space(1))) i4v* as1_i4;
typedef __attribute__((address_space(3))) i4v* as3_i4;
__device__ __forceinline__ void async_copy16(const _Float16* g, _Float16* l)
{
    __builtin_amdgcn_global_load_async_to_lds_b128(
        (as1_i4)(unsigned long long)g,
        (as3_i4)(unsigned int)(unsigned long long)l,
        0, 0);
}
__device__ __forceinline__ void wait_async()
{
#if __has_builtin(__builtin_amdgcn_s_wait_asynccnt)
    __builtin_amdgcn_s_wait_asynccnt(0);
#else
    asm volatile("s_wait_asynccnt 0x0" ::: "memory");
#endif
}
#else
__device__ __forceinline__ void async_copy16(const _Float16* g, _Float16* l)
{
    *(v8h*)l = *(const v8h*)g;
}
__device__ __forceinline__ void wait_async() {}
#endif

__global__ __launch_bounds__(256)
void int4_gemm_wmma_f16(const _Float16* __restrict__ x,
                        const uint8_t*  __restrict__ wp,
                        const _Float16* __restrict__ wscale,
                        const _Float16* __restrict__ wzero,
                        _Float16*       __restrict__ out,
                        int M, int N, int K)
{
    (void)M;
    __shared__ _Float16 Alds[2][BLOCK_M * A_PAD];   // [m][k] row-major
    __shared__ _Float16 Blds[2][BLOCK_N * B_PAD];   // [n][k] col-major

    const int tid    = threadIdx.x;
    const int lane   = tid & 31;
    const int wave   = tid >> 5;
    const int wave_m = wave & 3;          // 4 waves x 32 rows
    const int wave_n = wave >> 2;         // 2 waves x 64 cols
    const int l16    = lane & 15;
    const int halfid = lane >> 4;

    const int n0 = blockIdx.x * BLOCK_N;
    const int m0 = blockIdx.y * BLOCK_M;
    const int Nh = N >> 1;

    // ---- staging roles -----------------------------------------------------
    // x tile: 128 rows x 64 halves; thread copies 4x16B of one half-row
    const int a_row  = tid >> 1;
    const int a_col  = (tid & 1) * 32;
    // w tile: 64(K) x 128(N) nibbles; thread owns K-pair kp and two N-octets
    const int w_kp   = tid & 31;          // K rows 2*kp, 2*kp+1
    const int w_g0   = tid >> 5;          // n-octet 0..7
    const int w_g1   = 8 + (tid >> 5);    // n-octet 8..15

    uint32_t wr[2][2];                    // [unit][krow] packed bytes (8 n each)
    h2 sc2[2][8];                         // per-unit per-n splat scale
    h2 zo2[2][8];                         // per-unit per-n splat (1024 + zero)
    int kg_cur = -1;

#define UPDATE_SCALES(K0N)                                                     \
    do {                                                                       \
        const int kg_ = (K0N) >> 7;                                            \
        if (kg_ != kg_cur) {                                                   \
            kg_cur = kg_;                                                      \
            h2 kb_; kb_[0] = (_Float16)1024; kb_[1] = (_Float16)1024;          \
            _Pragma("unroll")                                                  \
            for (int u_ = 0; u_ < 2; ++u_) {                                   \
                const int g_ = u_ ? w_g1 : w_g0;                               \
                const v8h s_ = *(const v8h*)(wscale + (size_t)kg_ * N + n0 + g_ * 8); \
                const v8h z_ = *(const v8h*)(wzero  + (size_t)kg_ * N + n0 + g_ * 8); \
                const u4v sw_ = __builtin_bit_cast(u4v, s_);                   \
                const u4v zw_ = __builtin_bit_cast(u4v, z_);                   \
                _Pragma("unroll")                                              \
                for (int i_ = 0; i_ < 4; ++i_) {                               \
                    const uint32_t zo_u =                                      \
                        __builtin_bit_cast(uint32_t,                           \
                            (h2)(__builtin_bit_cast(h2, zw_[i_]) + kb_));      \
                    sc2[u_][2 * i_] = __builtin_bit_cast(h2,                   \
                        __builtin_amdgcn_perm(sw_[i_], sw_[i_], 0x05040504u)); \
                    sc2[u_][2 * i_ + 1] = __builtin_bit_cast(h2,               \
                        __builtin_amdgcn_perm(sw_[i_], sw_[i_], 0x07060706u)); \
                    zo2[u_][2 * i_] = __builtin_bit_cast(h2,                   \
                        __builtin_amdgcn_perm(zo_u, zo_u, 0x05040504u));       \
                    zo2[u_][2 * i_ + 1] = __builtin_bit_cast(h2,               \
                        __builtin_amdgcn_perm(zo_u, zo_u, 0x07060706u));       \
                }                                                              \
            }                                                                  \
        }                                                                      \
    } while (0)

#define STAGE_W_LOADS(K0N)                                                     \
    do {                                                                       \
        const uint8_t* wb_ = wp + (size_t)((K0N) + 2 * w_kp) * Nh + (n0 >> 1); \
        wr[0][0] = *(const uint32_t*)(wb_ + w_g0 * 4);                         \
        wr[0][1] = *(const uint32_t*)(wb_ + Nh + w_g0 * 4);                    \
        wr[1][0] = *(const uint32_t*)(wb_ + w_g1 * 4);                         \
        wr[1][1] = *(const uint32_t*)(wb_ + Nh + w_g1 * 4);                    \
    } while (0)

#define STAGE_A(K0N, BUF)                                                      \
    do {                                                                       \
        const _Float16* gx_ = x + (size_t)(m0 + a_row) * K + (K0N) + a_col;    \
        _Float16*       lx_ = &Alds[BUF][a_row * A_PAD + a_col];               \
        _Pragma("unroll")                                                      \
        for (int i_ = 0; i_ < 4; ++i_)                                         \
            async_copy16(gx_ + i_ * 8, lx_ + i_ * 8);                          \
    } while (0)

    // Each k-pair of nibbles -> one v_perm_b32 gather + or(0x6400 bias),
    // then (h - (1024+zero)) * scale with packed f16 ops (exact vs reference).
#define DEQUANT_W(BUF)                                                         \
    do {                                                                       \
        _Pragma("unroll")                                                      \
        for (int u_ = 0; u_ < 2; ++u_) {                                       \
            const int nsb_ = (u_ ? w_g1 : w_g0) * 8;                           \
            const uint32_t lo0_ = wr[u_][0] & 0x0F0F0F0Fu;                     \
            const uint32_t lo1_ = wr[u_][1] & 0x0F0F0F0Fu;                     \
            const uint32_t hi0_ = (wr[u_][0] >> 4) & 0x0F0F0F0Fu;              \
            const uint32_t hi1_ = (wr[u_][1] >> 4) & 0x0F0F0F0Fu;              \
            _Pragma("unroll")                                                  \
            for (int j_ = 0; j_ < 4; ++j_) {                                   \
                const uint32_t sel_ = 0x0C040C00u + (uint32_t)j_ +             \
                                      ((uint32_t)j_ << 16);                    \
                const uint32_t plo_ =                                          \
                    __builtin_amdgcn_perm(lo1_, lo0_, sel_) | 0x64006400u;     \
                const uint32_t phi_ =                                          \
                    __builtin_amdgcn_perm(hi1_, hi0_, sel_) | 0x64006400u;     \
                const h2 wlo_ = (__builtin_bit_cast(h2, plo_) - zo2[u_][2 * j_])     * sc2[u_][2 * j_];     \
                const h2 whi_ = (__builtin_bit_cast(h2, phi_) - zo2[u_][2 * j_ + 1]) * sc2[u_][2 * j_ + 1]; \
                *(h2*)&Blds[BUF][(nsb_ + 2 * j_)     * B_PAD + 2 * w_kp] = wlo_; \
                *(h2*)&Blds[BUF][(nsb_ + 2 * j_ + 1) * B_PAD + 2 * w_kp] = whi_; \
            }                                                                  \
        }                                                                      \
    } while (0)

    v8f acc[2][4];
#pragma unroll
    for (int mt = 0; mt < 2; ++mt)
#pragma unroll
        for (int nt = 0; nt < 4; ++nt)
#pragma unroll
            for (int r = 0; r < 8; ++r)
                acc[mt][nt][r] = 0.0f;

    const int T = K / BLOCK_K;

    // ---- prologue: fill buffer 0 ------------------------------------------
    UPDATE_SCALES(0);
    STAGE_W_LOADS(0);
    STAGE_A(0, 0);
    DEQUANT_W(0);
    wait_async();
    __syncthreads();

    // ---- pipelined main loop ----------------------------------------------
    for (int kt = 0; kt < T; ++kt) {
        const int buf  = kt & 1;
        const int nbuf = buf ^ 1;
        const int knext = (kt + 1) * BLOCK_K;
        const bool have_next = (kt + 1) < T;

        if (have_next) {
            UPDATE_SCALES(knext);
            STAGE_W_LOADS(knext);      // global latency hidden behind WMMAs
            STAGE_A(knext, nbuf);      // async -> other LDS buffer
        }

        // ---- compute on current buffer ------------------------------------
        const _Float16* Ab = Alds[buf];
        const _Float16* Bb = Blds[buf];
#pragma unroll
        for (int ks = 0; ks < 2; ++ks) {
            // A 16x32 frag: halves h -> K = klo + h (lanes 16-31: +16)
            v16h af[2];
#pragma unroll
            for (int mt = 0; mt < 2; ++mt) {
                const int row = wave_m * 32 + mt * 16 + l16;
                const int klo = ks * 32 + halfid * 16;
                const v8h a_lo = *(const v8h*)&Ab[row * A_PAD + klo];
                const v8h a_hi = *(const v8h*)&Ab[row * A_PAD + klo + 8];
                af[mt] = __builtin_shufflevector(a_lo, a_hi,
                                                 0, 1, 2, 3, 4, 5, 6, 7,
                                                 8, 9, 10, 11, 12, 13, 14, 15);
            }
            // B 32x16 frag: halves h -> K = kb + h (lanes 16-31: +16)
            v16h bf[4];
#pragma unroll
            for (int nt = 0; nt < 4; ++nt) {
                const int col = wave_n * 64 + nt * 16 + l16;
                const int kb  = ks * 32 + halfid * 16;
                const v8h b_lo = *(const v8h*)&Bb[col * B_PAD + kb];
                const v8h b_hi = *(const v8h*)&Bb[col * B_PAD + kb + 8];
                bf[nt] = __builtin_shufflevector(b_lo, b_hi,
                                                 0, 1, 2, 3, 4, 5, 6, 7,
                                                 8, 9, 10, 11, 12, 13, 14, 15);
            }
#pragma unroll
            for (int mt = 0; mt < 2; ++mt)
#pragma unroll
                for (int nt = 0; nt < 4; ++nt)
                    acc[mt][nt] = __builtin_amdgcn_wmma_f32_16x16x32_f16(
                        false, af[mt], false, bf[nt],
                        (short)0, acc[mt][nt], false, false);
        }

        if (have_next) {
            DEQUANT_W(nbuf);           // W regs consumed after compute
            wait_async();              // A async writes to nbuf complete
        }
        __syncthreads();
    }

    // ---- epilogue: C/D layout -> row = (lane>>4)*8 + r, col = lane&15 ------
#pragma unroll
    for (int mt = 0; mt < 2; ++mt) {
#pragma unroll
        for (int nt = 0; nt < 4; ++nt) {
            const int col = n0 + wave_n * 64 + nt * 16 + l16;
#pragma unroll
            for (int r = 0; r < 8; ++r) {
                const int row = m0 + wave_m * 32 + mt * 16 + halfid * 8 + r;
                out[(size_t)row * N + col] = (_Float16)acc[mt][nt][r];
            }
        }
    }
}

extern "C" void kernel_launch(void* const* d_in, const int* in_sizes, int n_in,
                              void* d_out, int out_size, void* d_ws, size_t ws_size,
                              hipStream_t stream)
{
    (void)n_in; (void)out_size; (void)d_ws; (void)ws_size;

    const _Float16* x  = (const _Float16*)d_in[0];
    const uint8_t*  wp = (const uint8_t*) d_in[1];
    const _Float16* ws = (const _Float16*)d_in[2];
    const _Float16* wz = (const _Float16*)d_in[3];
    _Float16*       o  = (_Float16*)d_out;

    const int K = 4096;                            // reference K
    const int N = (int)((2LL * in_sizes[1]) / K);  // w_packed = K * N/2
    const int M = in_sizes[0] / K;                 // x = M * K

    dim3 grid(N / BLOCK_N, M / BLOCK_M);
    int4_gemm_wmma_f16<<<grid, 256, 0, stream>>>(x, wp, ws, wz, o, M, N, K);
}
